// SYGNN_64433099374624
// MI455X (gfx1250) — compile-verified, matrix operand order checked by reading, not run on hardware
//
#include <hip/hip_runtime.h>
#include <math.h>

#define NATOMS 4096      // B*N = 8*512
#define NNB    64
#define NG     25
#define FEAT   128
#define CUTOFF_F 5.0f

typedef __attribute__((ext_vector_type(16))) __bf16 v16bf;
typedef __attribute__((ext_vector_type(8)))  float  v8f;

// ---- helpers -------------------------------------------------------------

__device__ __forceinline__ unsigned short f2bf(float x) {
  union { float f; unsigned u; } a; a.f = x;
  unsigned r = a.u + 0x7FFFu + ((a.u >> 16) & 1u);   // round-to-nearest-even
  return (unsigned short)(r >> 16);
}
__device__ __forceinline__ unsigned packbf(float lo, float hi) {
  return (unsigned)f2bf(lo) | ((unsigned)f2bf(hi) << 16);
}
__device__ __forceinline__ float softplus_f(float x) {
  return (x > 20.f) ? x : log1pf(__expf(x));
}

// dword index inside a 16-bit A/B fragment row (ISA 7.12.2: K halves interleaved)
__device__ __forceinline__ int frag_dw(int v, int half) {
  return (v < 4) ? (v + 4 * half) : (8 + (v - 4) + 4 * half);
}

// Load one 16x32 bf16 A/B fragment; `row` points at this lane's matrix row.
__device__ __forceinline__ v16bf load_frag(const unsigned* row, int kdw0, int half) {
  union { v16bf v; unsigned u[8]; } f;
#pragma unroll
  for (int i = 0; i < 8; ++i) f.u[i] = row[kdw0 + frag_dw(i, half)];
  return f.v;
}

__device__ __forceinline__ v8f wmma_bf16(v16bf a, v16bf b, v8f c) {
  return __builtin_amdgcn_wmma_f32_16x16x32_bf16(false, a, false, b, (short)0, c,
                                                 false, false);
}

// ---- kernel 1: weight transpose + bf16 convert ---------------------------

__global__ void k_convert(const float* wf1, const float* wf2, const float* w_in2f,
                          const float* w_f2out, const float* w_dense,
                          unsigned short* wf1T, unsigned short* wf2T,
                          unsigned short* in2fT, unsigned short* f2outT,
                          unsigned short* denseT) {
  int n = blockIdx.x;    // output row = original column, 0..127
  int k = threadIdx.x;   // 0..127
  if (k < 32) wf1T[n * 32 + k] = (k < NG) ? f2bf(wf1[k * FEAT + n]) : (unsigned short)0;
  wf2T  [n * FEAT + k] = f2bf(wf2   [k * FEAT + n]);
  in2fT [n * FEAT + k] = f2bf(w_in2f[k * FEAT + n]);
  f2outT[n * FEAT + k] = f2bf(w_f2out[k * FEAT + n]);
  denseT[n * FEAT + k] = f2bf(w_dense[k * FEAT + n]);
}

// ---- kernel 2: x = emb[Z] (write to out) ; y = x @ w_in2f ----------------

__global__ void k_embed_y(const float* emb, const int* az,
                          const unsigned short* in2fT, float* out, float* y) {
  __shared__ unsigned sX[64 * 64];           // 64 rows x 128 bf16
  int tid = threadIdx.x, lane = tid & 31, w = tid >> 5;
  int rowbase = blockIdx.x * 64;
  for (int idx = tid; idx < 64 * 128; idx += 128) {
    int rr = idx >> 7, cc = idx & 127;
    int atom = rowbase + rr;
    float v = emb[az[atom] * FEAT + cc];
    out[(long)atom * 256 + cc] = v;                      // first half of concat
    ((unsigned short*)sX)[rr * 128 + cc] = f2bf(v);
  }
  __syncthreads();
  int half = lane >> 4, lc = lane & 15;
  const unsigned* BT = (const unsigned*)in2fT;

  // A fragments depend only on ks -> load once, reuse across all 8 n-tiles
  v16bf afr[4];
#pragma unroll
  for (int ks = 0; ks < 4; ++ks)
    afr[ks] = load_frag(sX + (w * 16 + lc) * 64, ks * 16, half);

  for (int nt = 0; nt < 8; ++nt) {
    v8f c = {};
    const unsigned* brow = BT + (nt * 16 + lc) * 64;
#pragma unroll
    for (int ks = 0; ks < 4; ++ks)
      c = wmma_bf16(afr[ks], load_frag(brow, ks * 16, half), c);
#pragma unroll
    for (int r = 0; r < 8; ++r)
      y[(long)(rowbase + w * 16 + r + 8 * half) * FEAT + nt * 16 + lc] = c[r];
  }
}

// ---- kernel 3: fused edge pipeline ---------------------------------------
// block = 128 threads = 4 waves; wave w owns atom (blockIdx*4 + w), 64 edges.

__global__ void k_edge(const float* positions, const float* cell,
                       const float* cell_offset, const float* nmask,
                       const int* neighbors, const unsigned short* wf1T,
                       const float* bf1, const unsigned short* wf2T,
                       const float* bf2, const float* y, float* agg) {
  __shared__ unsigned sWf1T[128 * 16];        // 8 KB  (128 rows x 32 bf16)
  __shared__ unsigned sWf2T[128 * 64];        // 32 KB (128 rows x 128 bf16)
  __shared__ unsigned sFij [4 * 64 * 16];     // 16 KB (256 edges x 32 bf16)
  __shared__ unsigned short sW1[4 * 64 * 128];// 64 KB (256 edges x 128 bf16)
  __shared__ float sC[4 * 64];
  __shared__ int   sJ[4 * 64];

  int tid = threadIdx.x, lane = tid & 31, w = tid >> 5;

  for (int i = tid; i < 128 * 16; i += 128) sWf1T[i] = ((const unsigned*)wf1T)[i];
  for (int i = tid; i < 128 * 64; i += 128) sWf2T[i] = ((const unsigned*)wf2T)[i];

  int flat = blockIdx.x * 4 + w;
  int b = flat >> 9, n = flat & 511;
  const float* posb = positions + (long)b * 512 * 3;
  const float* cb   = cell + b * 9;
  float pix = posb[n * 3 + 0], piy = posb[n * 3 + 1], piz = posb[n * 3 + 2];
  const float step  = CUTOFF_F / (float)(NG - 1);
  const float coeff = -0.5f / (step * step);

  // phase 1: distances + Gaussian smearing (2 edges / lane)
  for (int e = lane; e < NNB; e += 32) {
    long eidx = ((long)(b * 512 + n)) * NNB + e;
    int   j = neighbors[eidx];
    float m = nmask[eidx];
    const float* co = cell_offset + eidx * 3;
    float ox = co[0] * cb[0] + co[1] * cb[3] + co[2] * cb[6];
    float oy = co[0] * cb[1] + co[1] * cb[4] + co[2] * cb[7];
    float oz = co[0] * cb[2] + co[1] * cb[5] + co[2] * cb[8];
    float dx = posb[j * 3 + 0] + ox - pix;
    float dy = posb[j * 3 + 1] + oy - piy;
    float dz = posb[j * 3 + 2] + oz - piz;
    float d2 = dx * dx + dy * dy + dz * dz;
    float r  = sqrtf(m > 0.f ? d2 : 1.0f) * m;
    sC[w * 64 + e] = ((r < CUTOFF_F) ? 1.f : 0.f) * m;
    sJ[w * 64 + e] = j;
    float g[26];
#pragma unroll
    for (int gi = 0; gi < NG; ++gi) { float t = r - gi * step; g[gi] = __expf(coeff * t * t); }
    g[25] = 0.f;
    unsigned u[16];
#pragma unroll
    for (int dwi = 0; dwi < 13; ++dwi) u[dwi] = packbf(g[2 * dwi], g[2 * dwi + 1]);
    u[13] = 0u; u[14] = 0u; u[15] = 0u;
    uint4* frow4 = (uint4*)(sFij + (w * 64 + e) * 16);   // 64B aligned
    frow4[0] = make_uint4(u[0],  u[1],  u[2],  u[3]);
    frow4[1] = make_uint4(u[4],  u[5],  u[6],  u[7]);
    frow4[2] = make_uint4(u[8],  u[9],  u[10], u[11]);
    frow4[3] = make_uint4(u[12], u[13], u[14], u[15]);
  }
  __syncthreads();

  int half = lane >> 4, lc = lane & 15;

  // phase 2: GEMM1 [64x32]@[32x128] (single K-step) + bias + softplus -> bf16 W1
  // B fragments depend only on nt -> load all 8 once, reuse across 4 m-tiles.
  {
    v16bf bfr1[8];
#pragma unroll
    for (int nt = 0; nt < 8; ++nt)
      bfr1[nt] = load_frag(sWf1T + (nt * 16 + lc) * 16, 0, half);

    for (int mt = 0; mt < 4; ++mt) {
      v16bf a = load_frag(sFij + (w * 64 + mt * 16 + lc) * 16, 0, half);
#pragma unroll
      for (int nt = 0; nt < 8; ++nt) {
        v8f c = {};
        c = wmma_bf16(a, bfr1[nt], c);
        float bias = bf1[nt * 16 + lc];
#pragma unroll
        for (int r = 0; r < 8; ++r) {
          int e = mt * 16 + r + 8 * half;
          sW1[(w * 64 + e) * 128 + nt * 16 + lc] = f2bf(softplus_f(c[r] + bias));
        }
      }
    }
  }
  __syncthreads();

  // phase 3: GEMM2 [64x128]@[128x128] + bias, cutoff, y-gather, edge reduction
  const unsigned* W1d = (const unsigned*)sW1;
  for (int nt = 0; nt < 8; ++nt) {
    int col = nt * 16 + lc;
    // B fragments (wf2^T) depend only on (nt, ks) -> hoist across m-tiles
    v16bf bfr[4];
#pragma unroll
    for (int ks = 0; ks < 4; ++ks)
      bfr[ks] = load_frag(sWf2T + col * 64, ks * 16, half);
    float b2 = bf2[col];
    float acc = 0.f;
    for (int mt = 0; mt < 4; ++mt) {
      v8f c = {};
      const unsigned* arow = W1d + (w * 64 + mt * 16 + lc) * 64;
#pragma unroll
      for (int ks = 0; ks < 4; ++ks)
        c = wmma_bf16(load_frag(arow, ks * 16, half), bfr[ks], c);
      // 8 consecutive edges for this lane: contiguous sC/sJ -> vector ds loads
      const float* cbase = &sC[w * 64 + mt * 16 + 8 * half];
      const int*   jbase = &sJ[w * 64 + mt * 16 + 8 * half];
#pragma unroll
      for (int r = 0; r < 8; ++r) {
        float w2 = (c[r] + b2) * cbase[r];
        acc += w2 * y[((long)(b * 512 + jbase[r])) * FEAT + col];
      }
    }
    acc += __shfl_xor(acc, 16, 32);   // lanes l and l+16 hold the same column
    if (lane < 16) agg[((long)(b * 512 + n)) * FEAT + col] = acc;
  }
}

// ---- kernel 4: v = (softplus(agg@w_f2out + b)) @ w_dense + b -------------

__global__ void k_tail(const float* agg, const unsigned short* f2outT,
                       const float* b_f2out, const unsigned short* denseT,
                       const float* b_dense, float* vout) {
  __shared__ unsigned sA[64 * 64];
  __shared__ unsigned sT[64 * 64];
  int tid = threadIdx.x, lane = tid & 31, w = tid >> 5;
  int rowbase = blockIdx.x * 64;
  for (int idx = tid; idx < 64 * 64; idx += 128) {
    int rr = idx >> 6, cc2 = idx & 63;
    const float* src = agg + (long)(rowbase + rr) * FEAT + cc2 * 2;
    sA[rr * 64 + cc2] = packbf(src[0], src[1]);
  }
  __syncthreads();
  int half = lane >> 4, lc = lane & 15;

  {
    const unsigned* BT1 = (const unsigned*)f2outT;
    v16bf afr[4];
#pragma unroll
    for (int ks = 0; ks < 4; ++ks)
      afr[ks] = load_frag(sA + (w * 16 + lc) * 64, ks * 16, half);
    for (int nt = 0; nt < 8; ++nt) {
      v8f c = {};
      const unsigned* brow = BT1 + (nt * 16 + lc) * 64;
#pragma unroll
      for (int ks = 0; ks < 4; ++ks)
        c = wmma_bf16(afr[ks], load_frag(brow, ks * 16, half), c);
      float bias = b_f2out[nt * 16 + lc];
#pragma unroll
      for (int r = 0; r < 8; ++r) {
        int m = w * 16 + r + 8 * half;
        ((unsigned short*)sT)[m * 128 + nt * 16 + lc] = f2bf(softplus_f(c[r] + bias));
      }
    }
  }
  __syncthreads();
  {
    const unsigned* BT2 = (const unsigned*)denseT;
    v16bf afr[4];
#pragma unroll
    for (int ks = 0; ks < 4; ++ks)
      afr[ks] = load_frag(sT + (w * 16 + lc) * 64, ks * 16, half);
    for (int nt = 0; nt < 8; ++nt) {
      v8f c = {};
      const unsigned* brow = BT2 + (nt * 16 + lc) * 64;
#pragma unroll
      for (int ks = 0; ks < 4; ++ks)
        c = wmma_bf16(afr[ks], load_frag(brow, ks * 16, half), c);
      float bias = b_dense[nt * 16 + lc];
#pragma unroll
      for (int r = 0; r < 8; ++r)
        vout[(long)(rowbase + w * 16 + r + 8 * half) * FEAT + nt * 16 + lc] = c[r] + bias;
    }
  }
}

// ---- kernel 5: symmetric gather into second half of concat ---------------

__global__ void k_gather(const float* v, const int* label, float* out) {
  int idx = blockIdx.x * 256 + threadIdx.x;   // NATOMS*128 elements
  int atom = idx >> 7, d = idx & 127;
  int bb = atom >> 9;
  int ln = label[atom];
  out[(long)atom * 256 + 128 + d] = v[((long)(bb * 512 + ln)) * FEAT + d];
}

// ---- launcher ------------------------------------------------------------

extern "C" void kernel_launch(void* const* d_in, const int* in_sizes, int n_in,
                              void* d_out, int out_size, void* d_ws, size_t ws_size,
                              hipStream_t stream) {
  const float* positions   = (const float*)d_in[0];
  const float* cell        = (const float*)d_in[1];
  const float* cell_offset = (const float*)d_in[2];
  const float* nmask       = (const float*)d_in[3];
  const float* emb         = (const float*)d_in[4];
  const float* wf1         = (const float*)d_in[5];
  const float* bf1         = (const float*)d_in[6];
  const float* wf2         = (const float*)d_in[7];
  const float* bf2         = (const float*)d_in[8];
  const float* w_in2f      = (const float*)d_in[9];
  const float* w_f2out     = (const float*)d_in[10];
  const float* b_f2out     = (const float*)d_in[11];
  const float* w_dense     = (const float*)d_in[12];
  const float* b_dense     = (const float*)d_in[13];
  const int*   az          = (const int*)d_in[14];
  const int*   neighbors   = (const int*)d_in[15];
  const int*   label       = (const int*)d_in[16];

  char* ws = (char*)d_ws;
  unsigned short* wf1T   = (unsigned short*)(ws + 0);        // 8 KB
  unsigned short* wf2T   = (unsigned short*)(ws + 8192);     // 32 KB
  unsigned short* in2fT  = (unsigned short*)(ws + 40960);    // 32 KB
  unsigned short* f2outT = (unsigned short*)(ws + 73728);    // 32 KB
  unsigned short* denseT = (unsigned short*)(ws + 106496);   // 32 KB
  float* y   = (float*)(ws + 139264);                        // 2 MB
  float* agg = (float*)(ws + 139264 + 2097152);              // 2 MB
  float* v   = (float*)(ws + 139264 + 2 * 2097152);          // 2 MB
  float* out = (float*)d_out;

  hipLaunchKernelGGL(k_convert, dim3(128), dim3(128), 0, stream,
                     wf1, wf2, w_in2f, w_f2out, w_dense,
                     wf1T, wf2T, in2fT, f2outT, denseT);
  hipLaunchKernelGGL(k_embed_y, dim3(NATOMS / 64), dim3(128), 0, stream,
                     emb, az, in2fT, out, y);
  hipLaunchKernelGGL(k_edge, dim3(NATOMS / 4), dim3(128), 0, stream,
                     positions, cell, cell_offset, nmask, neighbors,
                     wf1T, bf1, wf2T, bf2, y, agg);
  hipLaunchKernelGGL(k_tail, dim3(NATOMS / 64), dim3(128), 0, stream,
                     agg, f2outT, b_f2out, denseT, b_dense, v);
  hipLaunchKernelGGL(k_gather, dim3(NATOMS * 128 / 256), dim3(256), 0, stream,
                     v, label, out);
}
